// TransformerDecoder_43018392437273
// MI455X (gfx1250) — compile-verified, matrix operand order checked by reading, not run on hardware
//
#include <hip/hip_runtime.h>

#define BB 32
#define LL 12
#define NN 207
#define EE 128
#define HH 8
#define BN (BB * NN)     // 6624
#define MM (BN * LL)     // 79488  (divisible by 128)
#define NLAY 2
#define BSTRIDE 136      // padded LDS row stride (halves): 272B = 17 banks, conflict-free

typedef __attribute__((ext_vector_type(16))) _Float16 v16h;
typedef __attribute__((ext_vector_type(8)))  _Float16 v8h;
typedef __attribute__((ext_vector_type(8)))  float    v8f;
typedef __attribute__((ext_vector_type(4)))  float    v4f;

__device__ __forceinline__ v8f wmma_f16(v16h a, v16h b, v8f c) {
  // D = A(16x32 f16) x B(32x16 f16) + C(16x16 f32)
  return __builtin_amdgcn_wmma_f32_16x16x32_f16(false, a, false, b, (short)0, c, false, false);
}

// Global -> LDS 16-byte stage (global_load_b128 + ds_store_b128).
__device__ __forceinline__ void stage16(const _Float16* g, _Float16* l) {
  *(v8h*)l = *(const v8h*)g;
}

// A-operand fragment from a row-major f32 row (convert to f16).
// Lane layout (ISA 7.12.2, 16-bit A 16x32): lanes0-15 K{0..7,16..23}, lanes16-31 {8..15,24..31}.
__device__ __forceinline__ v16h frag_f32(const float* __restrict__ p, int o1, int o2) {
  v4f a0 = *(const v4f*)(p + o1);
  v4f a1 = *(const v4f*)(p + o1 + 4);
  v4f a2 = *(const v4f*)(p + o2);
  v4f a3 = *(const v4f*)(p + o2 + 4);
  v16h f;
#pragma unroll
  for (int i = 0; i < 4; ++i) {
    f[i]      = (_Float16)a0[i];
    f[4 + i]  = (_Float16)a1[i];
    f[8 + i]  = (_Float16)a2[i];
    f[12 + i] = (_Float16)a3[i];
  }
  return f;
}

__device__ __forceinline__ v16h frag_f16(const _Float16* p, int o1, int o2) {
  v8h x = *(const v8h*)(p + o1);
  v8h y = *(const v8h*)(p + o2);
  v16h f;
#pragma unroll
  for (int i = 0; i < 8; ++i) { f[i] = x[i]; f[8 + i] = y[i]; }
  return f;
}

__device__ __forceinline__ float wave_sum(float v) {
#pragma unroll
  for (int o = 16; o > 0; o >>= 1) v += __shfl_xor(v, o, 32);
  return v;
}

// ---------------- weight pack: f32 [*,*] -> f16 transposed [N,K] ----------------
__global__ void k_pack(const float* __restrict__ src, _Float16* __restrict__ dst,
                       int Nn, int Kk, int sn, int sk, int off) {
  const int idx = blockIdx.x * 256 + threadIdx.x;
  if (idx >= Nn * Kk) return;
  const int n = idx / Kk, k = idx % Kk;
  dst[idx] = (_Float16)src[off + n * sn + k * sk];
}

// ---------------- GEMM 128x128: C[M,128] = A[M,128] @ W + bias (+resid) (relu?) -
// WT = W^T f16 [128,128]. Block: 8 waves, each wave a 16(M) x 64(N) tile.
// Weight tile (64 cols x 128 K) staged once per block into LDS; A rows prefetched.
__global__ void __launch_bounds__(256) k_gemm(
    const float* __restrict__ A, const _Float16* __restrict__ WT,
    const float* __restrict__ bias, const float* __restrict__ resid,
    float* __restrict__ C, int relu) {
  __shared__ __align__(16) _Float16 bt[64 * BSTRIDE];   // ~17 KB
  const int wave = threadIdx.x >> 5, lane = threadIdx.x & 31;
  const int r = lane & 15;
  const bool hi = lane >= 16;
  const int o1 = hi ? 8 : 0, o2 = hi ? 24 : 16;
  const int m0 = blockIdx.x * 128 + wave * 16;
  const int n0 = blockIdx.y * 64;
  const float* arow = A + (size_t)(m0 + r) * EE;
  __builtin_prefetch(arow, 0, 3);          // global_prefetch_b8: warm A row while staging B
  // cooperative stage of WT rows [n0, n0+64) into LDS
  for (int q = threadIdx.x; q < 64 * 16; q += 256) {
    const int rr = q >> 4, ck = (q & 15) * 8;
    stage16(WT + (size_t)(n0 + rr) * EE + ck, bt + rr * BSTRIDE + ck);
  }
  __syncthreads();
  v8f acc[4] = {};
  for (int kk = 0; kk < EE; kk += 32) {
    v16h af = frag_f32(arow + kk, o1, o2);
#pragma unroll
    for (int c = 0; c < 4; ++c) {
      acc[c] = wmma_f16(af, frag_f16(bt + (c * 16 + r) * BSTRIDE + kk, o1, o2), acc[c]);
    }
  }
#pragma unroll
  for (int c = 0; c < 4; ++c) {
    const int n = n0 + c * 16 + r;
    const float bv = bias ? bias[n] : 0.0f;
#pragma unroll
    for (int i = 0; i < 8; ++i) {
      const int m = m0 + i + (hi ? 8 : 0);
      float v = acc[c][i] + bv;
      if (resid) v += resid[(size_t)m * EE + n];
      if (relu) v = fmaxf(v, 0.0f);
      C[(size_t)m * EE + n] = v;
    }
  }
}

// ---------------- 3-tap local conv as gated GEMMs (causal-tri or SAME) ---------
// WT3 = three f16 [E,E] blocks (tap-major). Token t = bn*L + l; tap k reads row t+k-1.
// All 3 tap weight tiles staged in LDS (~52 KB of the 320 KB WGP pool).
__global__ void __launch_bounds__(256) k_conv3(
    const float* __restrict__ A, const _Float16* __restrict__ WT3,
    const float* __restrict__ bias, float* __restrict__ C, int use_tri) {
  __shared__ __align__(16) _Float16 bt[3 * 64 * BSTRIDE];
  const int wave = threadIdx.x >> 5, lane = threadIdx.x & 31;
  const int r = lane & 15;
  const bool hi = lane >= 16;
  const int o1 = hi ? 8 : 0, o2 = hi ? 24 : 16;
  const int m0 = blockIdx.x * 128 + wave * 16;
  const int n0 = blockIdx.y * 64;
  const int t = m0 + r;
  __builtin_prefetch(A + (size_t)t * EE, 0, 3);
  for (int q = threadIdx.x; q < 3 * 64 * 16; q += 256) {
    const int tap = q / (64 * 16);
    const int rr = (q >> 4) & 63;
    const int ck = (q & 15) * 8;
    stage16(WT3 + (size_t)tap * EE * EE + (size_t)(n0 + rr) * EE + ck,
            bt + tap * 64 * BSTRIDE + rr * BSTRIDE + ck);
  }
  __syncthreads();
  const int l = t % LL;
  v8f acc[4] = {};
#pragma unroll
  for (int tap = 0; tap < 3; ++tap) {
    const int j = l + tap - 1;
    bool ok = (j >= 0) && (j < LL);
    if (use_tri) ok = ok && ((j <= l) || (j < 1));   // tri = tril | [:, :OFFSET]
    const float* arow = A + (ptrdiff_t)(t + tap - 1) * EE;  // deref only if ok
    const _Float16* bb = bt + tap * 64 * BSTRIDE;
    for (int kk = 0; kk < EE; kk += 32) {
      v16h af = {};
      if (ok) af = frag_f32(arow + kk, o1, o2);
#pragma unroll
      for (int c = 0; c < 4; ++c) {
        acc[c] = wmma_f16(af, frag_f16(bb + (c * 16 + r) * BSTRIDE + kk, o1, o2), acc[c]);
      }
    }
  }
#pragma unroll
  for (int c = 0; c < 4; ++c) {
    const int n = n0 + c * 16 + r;
    const float bv = bias ? bias[n] : 0.0f;
#pragma unroll
    for (int i = 0; i < 8; ++i) {
      const int m = m0 + i + (hi ? 8 : 0);
      C[(size_t)m * EE + n] = acc[c][i] + bv;
    }
  }
}

// ---------------- fused FFN: Y = relu(X@W1+b1)@W2 + b2 + X, hidden in LDS -------
__global__ void __launch_bounds__(256) k_ffn(
    const float* __restrict__ X, const _Float16* __restrict__ W1T,
    const float* __restrict__ b1v, const _Float16* __restrict__ W2T,
    const float* __restrict__ b2v, float* __restrict__ Y) {
  __shared__ __align__(16) _Float16 hid[16 * 512];   // 16KB
  const int wave = threadIdx.x >> 5, lane = threadIdx.x & 31;
  const int r = lane & 15;
  const bool hi = lane >= 16;
  const int o1 = hi ? 8 : 0, o2 = hi ? 24 : 16;
  const int m0 = blockIdx.x * 16;
  const float* arow = X + (size_t)(m0 + r) * EE;
  __builtin_prefetch(arow, 0, 3);
  // stage 1: hidden[16,512]; each wave handles its own 4 n-subtiles (no cross-wave reuse)
  v8f acc[4] = {};
  for (int kk = 0; kk < EE; kk += 32) {
    v16h af = frag_f32(arow + kk, o1, o2);
#pragma unroll
    for (int c = 0; c < 4; ++c) {
      const _Float16* wrow = W1T + (size_t)(wave * 64 + c * 16 + r) * EE + kk;
      acc[c] = wmma_f16(af, frag_f16(wrow, o1, o2), acc[c]);
    }
  }
#pragma unroll
  for (int c = 0; c < 4; ++c) {
    const int n = wave * 64 + c * 16 + r;
    const float bb = b1v[n];
#pragma unroll
    for (int i = 0; i < 8; ++i) {
      const int m = i + (hi ? 8 : 0);
      hid[m * 512 + n] = (_Float16)fmaxf(acc[c][i] + bb, 0.0f);
    }
  }
  __syncthreads();
  // stage 2: out[16,128]; one 16-col subtile per wave, K=512 read from LDS
  v8f acc2 = {};
  const _Float16* hrow = hid + (size_t)r * 512;
  for (int kk = 0; kk < 512; kk += 32) {
    v16h af = frag_f16(hrow + kk, o1, o2);
    const _Float16* wrow = W2T + (size_t)(wave * 16 + r) * 512 + kk;
    acc2 = wmma_f16(af, frag_f16(wrow, o1, o2), acc2);
  }
  const int n = wave * 16 + r;
  const float bb = b2v[n];
#pragma unroll
  for (int i = 0; i < 8; ++i) {
    const int m = m0 + i + (hi ? 8 : 0);
    Y[(size_t)m * EE + n] = acc2[i] + bb + X[(size_t)m * EE + n];
  }
}

// ---------------- attention core: softmax(QK^T/4) V, per (bn,h,lq) thread ------
__global__ void k_attn(const float* __restrict__ Q, const float* __restrict__ Kv,
                       const float* __restrict__ V, float* __restrict__ O, int causal) {
  const int idx = blockIdx.x * 256 + threadIdx.x;
  if (idx >= BN * HH * LL) return;
  const int lq = idx % LL;
  const int h  = (idx / LL) % HH;
  const int bn = idx / (LL * HH);
  const int d0 = h * 16;
  const float* q = Q + (size_t)(bn * LL + lq) * EE + d0;
  float qr[16];
#pragma unroll
  for (int i = 0; i < 16; ++i) qr[i] = q[i];
  const int lmax = causal ? (lq + 1) : LL;
  float s[LL];
  float mx = -1e30f;
  for (int lk = 0; lk < lmax; ++lk) {
    const float* kr = Kv + (size_t)(bn * LL + lk) * EE + d0;
    float a = 0.0f;
#pragma unroll
    for (int i = 0; i < 16; ++i) a += qr[i] * kr[i];
    a *= 0.25f;                      // 1/sqrt(16)
    s[lk] = a;
    mx = fmaxf(mx, a);
  }
  float sum = 0.0f;
  for (int lk = 0; lk < lmax; ++lk) { float e = __expf(s[lk] - mx); s[lk] = e; sum += e; }
  const float inv = 1.0f / sum;
  float o[16] = {};
  for (int lk = 0; lk < lmax; ++lk) {
    const float* vr = V + (size_t)(bn * LL + lk) * EE + d0;
    const float w = s[lk] * inv;
#pragma unroll
    for (int i = 0; i < 16; ++i) o[i] += w * vr[i];
  }
  float* op = O + (size_t)(bn * LL + lq) * EE + d0;
#pragma unroll
  for (int i = 0; i < 16; ++i) op[i] = o[i];
}

// ---------------- LayerNorm over E=128, one wave32 per row ---------------------
__global__ void k_ln(const float* __restrict__ X, const float* __restrict__ g,
                     const float* __restrict__ b, float* __restrict__ Y, int Md) {
  const int t = blockIdx.x * 8 + (threadIdx.x >> 5);
  if (t >= Md) return;
  const int lane = threadIdx.x & 31;
  const float* x = X + (size_t)t * EE;
  float v[4];
#pragma unroll
  for (int i = 0; i < 4; ++i) v[i] = x[lane + 32 * i];
  float s = v[0] + v[1] + v[2] + v[3];
  s = wave_sum(s);
  const float mean = s * (1.0f / 128.0f);
  float q = 0.0f;
#pragma unroll
  for (int i = 0; i < 4; ++i) { float d = v[i] - mean; q += d * d; }
  q = wave_sum(q);
  const float inv = rsqrtf(q * (1.0f / 128.0f) + 1e-5f);
  float* y = Y + (size_t)t * EE;
#pragma unroll
  for (int i = 0; i < 4; ++i) {
    const int e = lane + 32 * i;
    y[e] = (v[i] - mean) * inv * g[e] + b[e];
  }
}

// ---------------- embedding + sinusoidal positional encoding ------------------
__global__ void k_embed(const float* __restrict__ x, const float* __restrict__ ew,
                        const float* __restrict__ eb, float* __restrict__ h) {
  const size_t idx = (size_t)blockIdx.x * 256 + threadIdx.x;
  if (idx >= (size_t)MM * EE) return;
  const int e = (int)(idx & (EE - 1));
  const int t = (int)(idx >> 7);
  const int l = t % LL;
  const int bn = t / LL;
  const int b = bn / NN, n = bn % NN;
  const float xv = x[(size_t)(b * LL + l) * NN + n];          // F_IN == 1
  const float ang = (float)l * __expf((float)(e & ~1) * (-9.210340371976184f / (float)EE));
  const float pe = (e & 1) ? __cosf(ang) : __sinf(ang);
  h[idx] = xv * ew[e] + eb[e] + pe;
}

// ---------------- final projection E->1, permute to [B,L,N,1] ------------------
__global__ void k_fc(const float* __restrict__ hf, const float* __restrict__ w,
                     const float* __restrict__ bv, float* __restrict__ out, int Md) {
  const int t = blockIdx.x * 8 + (threadIdx.x >> 5);
  if (t >= Md) return;
  const int lane = threadIdx.x & 31;
  const float* xr = hf + (size_t)t * EE;
  float s = 0.0f;
#pragma unroll
  for (int i = 0; i < 4; ++i) s += xr[lane + 32 * i] * w[lane + 32 * i];
  s = wave_sum(s);
  if (lane == 0) {
    const int l = t % LL, bn = t / LL, b = bn / NN, n = bn % NN;
    out[(size_t)(b * LL + l) * NN + n] = s + bv[0];
  }
}

// ===============================================================================
extern "C" void kernel_launch(void* const* d_in, const int* in_sizes, int n_in,
                              void* d_out, int out_size, void* d_ws, size_t ws_size,
                              hipStream_t stream) {
  (void)in_sizes; (void)n_in; (void)out_size; (void)ws_size;
  const float* x = (const float*)d_in[0];
  const float* enc[4] = {(const float*)d_in[1], (const float*)d_in[2],
                         (const float*)d_in[3], (const float*)d_in[4]};
  // d_in[5] = tgt_mask (tril); handled analytically via `causal` flag.
  const int P = 6;  // params leaves start (jax tree-flatten, sorted dict keys)
  auto F = [&](int i) -> const float* { return (const float*)d_in[i]; };
  const float* emb_b = F(P + 0);
  const float* emb_w = F(P + 1);
  const float* fcb   = F(P + 2);
  const float* fcw   = F(P + 3);

  const size_t ME = (size_t)MM * EE;
  float* b0  = (float*)d_ws;
  float* b1  = b0 + ME;
  float* wsE = b0 + 2 * ME;
  float* Qb  = b0 + 3 * ME;
  float* Kb  = b0 + 4 * ME;
  float* Vb  = b0 + 5 * ME;
  float* Tb  = b0 + 6 * ME;
  _Float16* w16 = (_Float16*)(b0 + 7 * ME);

  // ---- pack all weights to f16 transposed [N,K] ----
  size_t cur = 0;
  auto pack = [&](const float* src, int Nn, int Kk, int sn, int sk, int off) -> size_t {
    size_t o = cur;
    cur += (size_t)Nn * Kk;
    const int tot = Nn * Kk;
    k_pack<<<(tot + 255) / 256, 256, 0, stream>>>(src, w16 + o, Nn, Kk, sn, sk, off);
    return o;
  };
  auto packW = [&](const float* w, int Kk, int Nn) { return pack(w, Nn, Kk, 1, Nn, 0); };
  auto packConv = [&](const float* w) {  // w [E_out, E_in, 3] -> 3x f16 [E,E]
    size_t o = pack(w, EE, EE, EE * 3, 3, 0);
    pack(w, EE, EE, EE * 3, 3, 1);
    pack(w, EE, EE, EE * 3, 3, 2);
    return o;
  };

  struct CrossW {
    size_t wq, wk, wv, wo, w1, w2;
    const float *bq, *bk, *bv, *bo, *bb1, *bb2, *lag, *lab, *lfg, *lfb;
  };
  struct LayerW {
    size_t convq, convk[4], swq, swk, swv, swo;
    const float *ckb[4], *sbq, *sbk, *sbv, *sbo, *lsg, *lsb;
    CrossW cr[4];
  } lw[NLAY];

  for (int l = 0; l < NLAY; ++l) {
    const int LB = P + 4 + l * 83;
    for (int i = 0; i < 4; ++i) lw[l].ckb[i] = F(LB + i);            // convk_b[0..3]
    for (int i = 0; i < 4; ++i) lw[l].convk[i] = packConv(F(LB + 4 + i));
    lw[l].convq = packConv(F(LB + 8));
    for (int i = 0; i < 4; ++i) {                                     // cross[i]
      const int CB = LB + 9 + i * 16;  // bk,bo,bq,bv,wk,wo,wq,wv,b1,b2,lab,lag,lfb,lfg,w1,w2
      CrossW& c = lw[l].cr[i];
      c.bk = F(CB + 0); c.bo = F(CB + 1); c.bq = F(CB + 2); c.bv = F(CB + 3);
      c.wk = packW(F(CB + 4), EE, EE);
      c.wo = packW(F(CB + 5), EE, EE);
      c.wq = packW(F(CB + 6), EE, EE);
      c.wv = packW(F(CB + 7), EE, EE);
      c.bb1 = F(CB + 8); c.bb2 = F(CB + 9);
      c.lab = F(CB + 10); c.lag = F(CB + 11);
      c.lfb = F(CB + 12); c.lfg = F(CB + 13);
      c.w1 = packW(F(CB + 14), EE, 4 * EE);       // [128,512] -> WT [512,128]
      c.w2 = packW(F(CB + 15), 4 * EE, EE);       // [512,128] -> WT [128,512]
    }
    lw[l].lsb = F(LB + 73); lw[l].lsg = F(LB + 74);
    lw[l].sbk = F(LB + 75); lw[l].sbo = F(LB + 76);
    lw[l].sbq = F(LB + 77); lw[l].sbv = F(LB + 78);
    lw[l].swk = packW(F(LB + 79), EE, EE);
    lw[l].swo = packW(F(LB + 80), EE, EE);
    lw[l].swq = packW(F(LB + 81), EE, EE);
    lw[l].swv = packW(F(LB + 82), EE, EE);
  }

  // ---- compute pipeline ----
  auto gemm = [&](const float* A, size_t wt, const float* bias, const float* resid,
                  float* C, int relu) {
    dim3 grid(MM / 128, EE / 64);
    k_gemm<<<grid, 256, 0, stream>>>(A, w16 + wt, bias, resid, C, relu);
  };
  auto conv3 = [&](const float* A, size_t wt, const float* bias, float* C, int tri) {
    dim3 grid(MM / 128, EE / 64);
    k_conv3<<<grid, 256, 0, stream>>>(A, w16 + wt, bias, C, tri);
  };
  auto attn = [&](const float* Qp, const float* Kp, const float* Vp, float* Op, int causal) {
    k_attn<<<(BN * HH * LL) / 256, 256, 0, stream>>>(Qp, Kp, Vp, Op, causal);
  };
  auto ln = [&](const float* X, const float* g, const float* bv, float* Y) {
    k_ln<<<MM / 8, 256, 0, stream>>>(X, g, bv, Y, MM);
  };

  k_embed<<<(int)((ME + 255) / 256), 256, 0, stream>>>(x, emb_w, emb_b, b0);

  float* hcur = b0;   // current hidden state [M,E]
  float* haux = b1;
  for (int l = 0; l < NLAY; ++l) {
    LayerW& W = lw[l];
    // masked local causal conv: h2 = conv(h)
    conv3(hcur, W.convq, nullptr, haux, 1);
    // self attention on h2 (causal)
    gemm(haux, W.swq, W.sbq, nullptr, Qb, 0);
    gemm(haux, W.swk, W.sbk, nullptr, Kb, 0);
    gemm(haux, W.swv, W.sbv, nullptr, Vb, 0);
    attn(Qb, Kb, Vb, Tb, 1);
    gemm(Tb, W.swo, W.sbo, /*resid=*/haux, Qb, 0);
    ln(Qb, W.lsg, W.lsb, hcur);                         // h3
    for (int i = 0; i < 4; ++i) {
      CrossW& c = W.cr[i];
      conv3(enc[i], W.convk[i], W.ckb[i], wsE, 0);      // enc conv1d (SAME)
      gemm(hcur, c.wq, c.bq, nullptr, Qb, 0);
      gemm(wsE,  c.wk, c.bk, nullptr, Kb, 0);
      gemm(wsE,  c.wv, c.bv, nullptr, Vb, 0);
      attn(Qb, Kb, Vb, Tb, 0);
      gemm(Tb, c.wo, c.bo, /*resid=*/hcur, Qb, 0);
      ln(Qb, c.lag, c.lab, haux);                       // h4
      k_ffn<<<MM / 16, 256, 0, stream>>>(haux, w16 + c.w1, c.bb1,
                                         w16 + c.w2, c.bb2, Tb);     // f + h4
      ln(Tb, c.lfg, c.lfb, hcur);                       // h5
    }
  }
  k_fc<<<MM / 8, 256, 0, stream>>>(hcur, fcw, fcb, (float*)d_out, MM);
}
